// MultiHeadAttention_5368709120039
// MI455X (gfx1250) — matrix operand order repaired and checked by reading, not compiled, not run on hardware
//
#include <hip/hip_runtime.h>
#include <hip/hip_bf16.h>
#include <math.h>

// ---- problem constants (from reference) ----
#define DMODEL 1024
#define NHEADS 16
#define DK     64
#define BB     4
#define TT     2048
#define MM     (BB*TT)          // 8192 rows for projections

typedef __attribute__((ext_vector_type(16))) _Float16 v16h;
typedef __attribute__((ext_vector_type(8)))  _Float16 v8h;
typedef __attribute__((ext_vector_type(4)))  _Float16 v4h;
typedef __attribute__((ext_vector_type(8)))  float    v8f;
typedef __attribute__((ext_vector_type(4)))  float    v4f;
typedef __attribute__((ext_vector_type(4)))  int      v4i;

union F16x16 { v16h v; v8h h[2]; };

#define NEG_BIG (-3.0e38f)

// ---- gfx1250 async-LDS path (probe via __has_builtin, safe fallback) ----
#define AS_GLOBAL __attribute__((address_space(1)))
#define AS_LOCAL  __attribute__((address_space(3)))
#if defined(__has_builtin)
#  if __has_builtin(__builtin_amdgcn_global_load_async_to_lds_b128) && \
      __has_builtin(__builtin_amdgcn_s_wait_asynccnt)
#    define HAVE_ASYNC_LDS 1
#  endif
#endif
#ifndef HAVE_ASYNC_LDS
#  define HAVE_ASYNC_LDS 0
#endif

// ------------------------------------------------------------------
// fp32 -> f16 converter (for weights), 4 elems/thread
// ------------------------------------------------------------------
__global__ void __launch_bounds__(256)
cvt_f32_f16_kernel(const float* __restrict__ in,
                   _Float16* __restrict__ out, int n) {
    int i = (blockIdx.x * blockDim.x + threadIdx.x) * 4;
    if (i + 3 < n) {
        v4f f = *(const v4f*)(in + i);
        v4h h;
        #pragma unroll
        for (int j = 0; j < 4; ++j) h[j] = (_Float16)f[j];
        *(v4h*)(out + i) = h;
    }
}

// ------------------------------------------------------------------
// GEMM: Y = X @ W^T + bias.  X: [M=8192, K=1024], W(f16): [N=1024, K=1024].
// Block = 256 threads = 8 waves; each wave computes a 16x64 strip sharing
// one 64x32 W tile per K-step. W tile is staged into double-buffered LDS
// (gfx1250 async copy, ASYNCcnt-tracked), pipelined 1 step ahead.
// A_F16:   X is f16 (O-projection input) vs fp32 (converted on the fly).
// OUTMODE: 0 = f16 head-split [B,H,T,64] (Q,K)
//          1 = f16 head-split transposed [B,H,64,T] (V)
//          2 = f32 row-major [M,N] (final output)
// ------------------------------------------------------------------
template <bool A_F16, int OUTMODE>
__global__ void __launch_bounds__(256, 1)
gemm_wT_kernel(const void* __restrict__ Xv,
               const _Float16* __restrict__ W,
               const float* __restrict__ bias,
               void* __restrict__ Yv) {
    constexpr int K = DMODEL, N = DMODEL;
    const int tid  = threadIdx.x;
    const int lane = tid & 31;
    const int wave = tid >> 5;
    const int m0 = (blockIdx.x * 8 + wave) * 16;   // gridDim.x = M/128
    const int n0 = blockIdx.y * 64;                // gridDim.y = N/64
    const int row  = lane & 15;
    const int half = lane >> 4;

    const float*    Xf = (const float*)Xv;
    const _Float16* Xh = (const _Float16*)Xv;

    // double-buffered 64(cols) x 32(k) f16 W tile
    __shared__ __align__(16) _Float16 Wlds[2][64 * 32];

    // one b128 chunk per thread per stage: 64 rows * 4 chunks = 256
    const int wn  = tid >> 2;          // 0..63  (local col)
    const int wko = (tid & 3) * 8;     // 0,8,16,24 (k offset, halves)

    auto stageW = [&](int buf, int k0) {
        const _Float16* gsrc = W + (size_t)(n0 + wn) * K + k0 + wko;
        _Float16* ldst = &Wlds[buf][wn * 32 + wko];
#if HAVE_ASYNC_LDS
        __builtin_amdgcn_global_load_async_to_lds_b128(
            (AS_GLOBAL v4i*)gsrc, (AS_LOCAL v4i*)ldst, 0, 0);
#else
        *(v8h*)ldst = *(const v8h*)gsrc;
#endif
    };

    v8f acc[4] = {};
    const int akA = half * 8;    // A frag K offset within 32-chunk
    const int bkA = half * 8;    // B frag K offset within 32-chunk

    stageW(0, 0);
#if HAVE_ASYNC_LDS
    __builtin_amdgcn_s_wait_asynccnt(0);
#endif
    __syncthreads();

    int cur = 0;
    for (int k0 = 0; k0 < K; k0 += 32) {
        // ---- prefetch next W tile into the other LDS buffer ----
        if (k0 + 32 < K) stageW(cur ^ 1, k0 + 32);

        // ---- A fragment: 16x32 f16, lane=row, two 8-half runs ----
        F16x16 a;
        if (A_F16) {
            const _Float16* p = Xh + (size_t)(m0 + row) * K + k0 + akA;
            a.h[0] = *(const v8h*)(p);
            a.h[1] = *(const v8h*)(p + 16);
            if (k0 + 32 < K) __builtin_prefetch(p + 32, 0, 3);
        } else {
            const float* p = Xf + (size_t)(m0 + row) * K + k0 + akA;
            v4f f0 = *(const v4f*)(p);
            v4f f1 = *(const v4f*)(p + 4);
            v4f f2 = *(const v4f*)(p + 16);
            v4f f3 = *(const v4f*)(p + 20);
            if (k0 + 32 < K) __builtin_prefetch(p + 32, 0, 3);
            v8h lo, hi;
            #pragma unroll
            for (int i = 0; i < 4; ++i) {
                lo[i]     = (_Float16)f0[i];
                lo[4 + i] = (_Float16)f1[i];
                hi[i]     = (_Float16)f2[i];
                hi[4 + i] = (_Float16)f3[i];
            }
            a.h[0] = lo; a.h[1] = hi;
        }

        // ---- 4 B fragments from LDS (two 8-half runs per lane) ----
        #pragma unroll
        for (int t = 0; t < 4; ++t) {
            const _Float16* wp = &Wlds[cur][(t * 16 + row) * 32 + bkA];
            F16x16 b;
            b.h[0] = *(const v8h*)(wp);
            b.h[1] = *(const v8h*)(wp + 16);
            acc[t] = __builtin_amdgcn_wmma_f32_16x16x32_f16(
                false, a.v, false, b.v, (short)0, acc[t], false, false);
        }

#if HAVE_ASYNC_LDS
        __builtin_amdgcn_s_wait_asynccnt(0);
#endif
        __syncthreads();
        cur ^= 1;
    }

    // ---- epilogue: bias + store in requested layout ----
    #pragma unroll
    for (int t = 0; t < 4; ++t) {
        const int n = n0 + t * 16 + (lane & 15);
        const float bv = bias[n];
        #pragma unroll
        for (int r = 0; r < 8; ++r) {
            const int m = m0 + half * 8 + r;
            const float val = acc[t][r] + bv;
            if (OUTMODE == 2) {
                ((float*)Yv)[(size_t)m * N + n] = val;
            } else {
                const int bb  = m >> 11;        // / TT
                const int tok = m & (TT - 1);
                const int h   = n >> 6;         // / DK
                const int d   = n & (DK - 1);
                _Float16* Y = (_Float16*)Yv;
                if (OUTMODE == 0)
                    Y[((size_t)(bb * NHEADS + h) * TT + tok) * DK + d] = (_Float16)val;
                else
                    Y[((size_t)(bb * NHEADS + h) * DK + d) * TT + tok] = (_Float16)val;
            }
        }
    }
}

// ------------------------------------------------------------------
// Flash attention, causal.  1 wave per (b,h, 16-query tile).
// Q,K: [B,H,T,64] f16; Vt: [B,H,64,T] f16; AO: [B,T,1024] f16.
// Per 32-key chunk:
//   - load ALL K fragments (one clause), 4 S-wmma
//   - issue V fragment loads immediately (overlap with softmax VALU)
//   - online softmax via shfl_xor row reductions
//   - P re-layout C->A through LDS, 4 P.V wmma
// __launch_bounds__(32,1): single wave, full VGPR budget -> no spills.
// ------------------------------------------------------------------
__global__ void __launch_bounds__(32, 1)
attn_kernel(const _Float16* __restrict__ Q,
            const _Float16* __restrict__ Kmat,
            const _Float16* __restrict__ Vt,
            _Float16* __restrict__ AO) {
    const int qt   = blockIdx.x;           // T/16 tiles
    const int bh   = blockIdx.y;           // B*H
    const int lane = threadIdx.x;
    const int col  = lane & 15;
    const int half = lane >> 4;
    const int q0   = qt * 16;

    const _Float16* Qh = Q    + (size_t)bh * TT * DK;
    const _Float16* Kh = Kmat + (size_t)bh * TT * DK;
    const _Float16* Vh = Vt   + (size_t)bh * DK * TT;

    __shared__ __align__(16) _Float16 Plds[16][40];   // 16x32 P tile, padded

    // Q A-fragments for the two 32-wide K(dim) chunks
    F16x16 aq[2];
    #pragma unroll
    for (int c = 0; c < 2; ++c) {
        const _Float16* p = Qh + (size_t)(q0 + col) * DK + c * 32 + half * 8;
        aq[c].h[0] = *(const v8h*)(p);
        aq[c].h[1] = *(const v8h*)(p + 16);
    }

    v8f o[4] = {};
    float mrow[8], lrow[8];
    #pragma unroll
    for (int r = 0; r < 8; ++r) { mrow[r] = NEG_BIG; lrow[r] = 0.0f; }

    const int jend = q0 + 16;                // last needed key + 1 (causal)
    const int nchunks = (jend + 31) / 32;
    const float scale = 0.125f;              // 1/sqrt(64)

    for (int ch = 0; ch < nchunks; ++ch) {
        const int j0 = ch * 32;

        // ---- load all K fragments for this chunk up front ----
        F16x16 bk[2][2];
        #pragma unroll
        for (int st = 0; st < 2; ++st)
            #pragma unroll
            for (int c = 0; c < 2; ++c) {
                const _Float16* kp =
                    Kh + (size_t)(j0 + st * 16 + col) * DK + c * 32 + half * 8;
                bk[st][c].h[0] = *(const v8h*)(kp);
                bk[st][c].h[1] = *(const v8h*)(kp + 16);
            }

        // ---- S = Q K^T : two 16x16 tiles over 32 keys ----
        v8f s[2] = {};
        #pragma unroll
        for (int st = 0; st < 2; ++st)
            #pragma unroll
            for (int c = 0; c < 2; ++c)
                s[st] = __builtin_amdgcn_wmma_f32_16x16x32_f16(
                    false, aq[c].v, false, bk[st][c].v, (short)0, s[st],
                    false, false);

        // ---- issue V fragment loads now; they overlap the softmax math ----
        F16x16 bv[4];
        #pragma unroll
        for (int t = 0; t < 4; ++t) {
            const _Float16* vp =
                Vh + (size_t)(t * 16 + col) * TT + j0 + half * 8;
            bv[t].h[0] = *(const v8h*)(vp);
            bv[t].h[1] = *(const v8h*)(vp + 16);
        }

        // ---- scale + causal mask (lane holds key-col, rows half*8+r) ----
        #pragma unroll
        for (int st = 0; st < 2; ++st) {
            const int kidx = j0 + st * 16 + col;
            #pragma unroll
            for (int r = 0; r < 8; ++r) {
                const int qidx = q0 + half * 8 + r;
                float v = s[st][r] * scale;
                if (kidx > qidx) v = NEG_BIG;
                s[st][r] = v;
            }
        }

        // ---- online softmax: row max / correction / row sum ----
        #pragma unroll
        for (int r = 0; r < 8; ++r) {
            float v = fmaxf(s[0][r], s[1][r]);
            #pragma unroll
            for (int mk = 8; mk >= 1; mk >>= 1)
                v = fmaxf(v, __shfl_xor(v, mk, 32));
            const float mnew = fmaxf(mrow[r], v);
            const float corr = __expf(mrow[r] - mnew);
            mrow[r] = mnew;
            const float p0 = __expf(s[0][r] - mnew);
            const float p1 = __expf(s[1][r] - mnew);
            s[0][r] = p0; s[1][r] = p1;
            float pr = p0 + p1;
            #pragma unroll
            for (int mk = 8; mk >= 1; mk >>= 1)
                pr += __shfl_xor(pr, mk, 32);
            lrow[r] = lrow[r] * corr + pr;
            #pragma unroll
            for (int t = 0; t < 4; ++t) o[t][r] *= corr;
        }

        // ---- P tile -> LDS (C-layout) ----
        #pragma unroll
        for (int st = 0; st < 2; ++st)
            #pragma unroll
            for (int r = 0; r < 8; ++r)
                Plds[half * 8 + r][st * 16 + col] = (_Float16)s[st][r];
        __syncthreads();

        // ---- reload P in A-fragment layout (16x32, lane=row) ----
        F16x16 ap;
        const _Float16* pp = &Plds[col][half * 8];
        ap.h[0] = *(const v8h*)(pp);
        ap.h[1] = *(const v8h*)(pp + 16);
        __syncthreads();

        // ---- O += P * V  (V transposed: column d is T-contiguous) ----
        #pragma unroll
        for (int t = 0; t < 4; ++t)
            o[t] = __builtin_amdgcn_wmma_f32_16x16x32_f16(
                false, ap.v, false, bv[t].v, (short)0, o[t], false, false);
    }

    // ---- finalize: O /= l ; store to [B,T,1024] f16 (heads concatenated) ----
    const int b = bh >> 4, h = bh & 15;
    #pragma unroll
    for (int r = 0; r < 8; ++r) {
        const float inv = 1.0f / lrow[r];
        const int tok = q0 + half * 8 + r;
        #pragma unroll
        for (int t = 0; t < 4; ++t) {
            AO[((size_t)(b * TT + tok)) * DMODEL + h * DK + t * 16 + col] =
                (_Float16)(o[t][r] * inv);
        }
    }
}

// ------------------------------------------------------------------
// Launch: weights->f16, QKV projections, flash attention, O projection.
// Workspace layout (bytes):
//   [0,2M)   Wq16   [2M,4M)  Wk16   [4M,6M)  Wv16   [6M,8M) Wo16
//   [8M,24M) Q16    [24M,40M) K16   [40M,56M) Vt16  [56M,72M) AO16
// ------------------------------------------------------------------
extern "C" void kernel_launch(void* const* d_in, const int* in_sizes, int n_in,
                              void* d_out, int out_size, void* d_ws, size_t ws_size,
                              hipStream_t stream) {
    const float* q  = (const float*)d_in[0];
    const float* k  = (const float*)d_in[1];
    const float* v  = (const float*)d_in[2];
    // d_in[3] = mask (causal, recomputed analytically)
    const float* Wq = (const float*)d_in[4];
    const float* bq = (const float*)d_in[5];
    const float* Wk = (const float*)d_in[6];
    const float* bk = (const float*)d_in[7];
    const float* Wv = (const float*)d_in[8];
    const float* bv = (const float*)d_in[9];
    const float* Wo = (const float*)d_in[10];
    const float* bo = (const float*)d_in[11];

    char* ws = (char*)d_ws;
    const size_t MB = 1024u * 1024u;
    _Float16* Wq16 = (_Float16*)(ws + 0 * MB);
    _Float16* Wk16 = (_Float16*)(ws + 2 * MB);
    _Float16* Wv16 = (_Float16*)(ws + 4 * MB);
    _Float16* Wo16 = (_Float16*)(ws + 6 * MB);
    _Float16* Q16  = (_Float16*)(ws + 8 * MB);
    _Float16* K16  = (_Float16*)(ws + 24 * MB);
    _Float16* Vt16 = (_Float16*)(ws + 40 * MB);
    _Float16* AO16 = (_Float16*)(ws + 56 * MB);

    const int nW = DMODEL * DMODEL;
    const int cvtBlocks = nW / (256 * 4);
    cvt_f32_f16_kernel<<<cvtBlocks, 256, 0, stream>>>(Wq, Wq16, nW);
    cvt_f32_f16_kernel<<<cvtBlocks, 256, 0, stream>>>(Wk, Wk16, nW);
    cvt_f32_f16_kernel<<<cvtBlocks, 256, 0, stream>>>(Wv, Wv16, nW);
    cvt_f32_f16_kernel<<<cvtBlocks, 256, 0, stream>>>(Wo, Wo16, nW);

    dim3 gblk(MM / 128, DMODEL / 64);   // (64, 16)
    gemm_wT_kernel<false, 0><<<gblk, 256, 0, stream>>>(q, Wq16, bq, Q16);
    gemm_wT_kernel<false, 0><<<gblk, 256, 0, stream>>>(k, Wk16, bk, K16);
    gemm_wT_kernel<false, 1><<<gblk, 256, 0, stream>>>(v, Wv16, bv, Vt16);

    attn_kernel<<<dim3(TT / 16, BB * NHEADS), 32, 0, stream>>>(Q16, K16, Vt16, AO16);

    gemm_wT_kernel<true, 2><<<gblk, 256, 0, stream>>>(AO16, Wo16, bo, d_out);
}
